// PredictiveCoding_37005438222933
// MI455X (gfx1250) — matrix operand order repaired and checked by reading, not compiled, not run on hardware
//
#include <hip/hip_runtime.h>

// Problem constants (from reference)
#define NN 8192
#define MM 8192
#define DD 512

typedef unsigned short u16;
typedef __attribute__((ext_vector_type(16))) __bf16 v16bf;
typedef __attribute__((ext_vector_type(8)))  float  v8f;
typedef __attribute__((ext_vector_type(4)))  float  v4f;

union FragU { v16bf v; v4f f4[2]; };

// ---- bf16 bit helpers (RNE) -------------------------------------------------
static __device__ __forceinline__ u16 f2bf(float x) {
  unsigned u = __float_as_uint(x);
  unsigned r = u + 0x7FFFu + ((u >> 16) & 1u);
  return (u16)(r >> 16);
}
static __device__ __forceinline__ float bf2f(u16 h) {
  return __uint_as_float(((unsigned)h) << 16);
}

// ---- WMMA fragment loads (CDNA5 16-bit layouts, ISA 7.12.2) -----------------
// A 16x32: lane<16 -> row, K = {k0..k0+7} in v0..3, {k0+16..k0+23} in v4..7
//          lane>=16 -> same row set, K offset +8
static __device__ __forceinline__ v16bf frag_A(const u16* rowp, int k0, int half) {
  FragU u;
  u.f4[0] = *reinterpret_cast<const v4f*>(rowp + k0 + 8 * half);
  u.f4[1] = *reinterpret_cast<const v4f*>(rowp + k0 + 16 + 8 * half);
  return u.v;
}
// B 32x16: lane = column (row of the row-major operand), same K order as A:
//          {k0..k0+7} in v0..3, {k0+16..k0+23} in v4..7, lanes>=16 K offset +8
static __device__ __forceinline__ v16bf frag_B(const u16* rowp, int k0, int half) {
  FragU u;
  u.f4[0] = *reinterpret_cast<const v4f*>(rowp + k0 + 8 * half);
  u.f4[1] = *reinterpret_cast<const v4f*>(rowp + k0 + 16 + 8 * half);
  return u.v;
}

// bf16x3 split product: acc += Ahi*Bhi + Ahi*Blo + Alo*Bhi
static __device__ __forceinline__ v8f wmma3(v16bf ah, v16bf al, v16bf bh, v16bf bl, v8f acc) {
  acc = __builtin_amdgcn_wmma_f32_16x16x32_bf16(false, ah, false, bh, (short)0, acc, false, false);
  acc = __builtin_amdgcn_wmma_f32_16x16x32_bf16(false, ah, false, bl, (short)0, acc, false, false);
  acc = __builtin_amdgcn_wmma_f32_16x16x32_bf16(false, al, false, bh, (short)0, acc, false, false);
  return acc;
}

// ---- Kernel 1: split fp32 rows into bf16 hi/lo + row sum of squares ---------
__global__ __launch_bounds__(256) void prep_split_kernel(
    const float* __restrict__ src, u16* __restrict__ hi, u16* __restrict__ lo,
    float* __restrict__ sq) {
  const int row = blockIdx.x;
  const float* p = src + (size_t)row * DD;
  float acc = 0.f;
  for (int c = threadIdx.x; c < DD; c += 256) {
    float v = p[c];
    u16 hb = f2bf(v);
    float hf = bf2f(hb);
    u16 lb = f2bf(v - hf);
    hi[(size_t)row * DD + c] = hb;
    lo[(size_t)row * DD + c] = lb;
    acc += v * v;
  }
  __shared__ float red[256];
  red[threadIdx.x] = acc;
  __syncthreads();
  for (int s = 128; s > 0; s >>= 1) {
    if (threadIdx.x < s) red[threadIdx.x] += red[threadIdx.x + s];
    __syncthreads();
  }
  if (threadIdx.x == 0) sq[row] = red[0];
}

// ---- Kernel 2: f = tanh(mu @ W^T + b), split-bf16 WMMA ----------------------
__global__ __launch_bounds__(256) void fmu_gemm_kernel(
    const u16* __restrict__ mu_hi, const u16* __restrict__ mu_lo,
    const u16* __restrict__ w_hi,  const u16* __restrict__ w_lo,
    const float* __restrict__ bias, u16* __restrict__ f_hi, u16* __restrict__ f_lo) {
  const int wid  = blockIdx.x * 8 + (threadIdx.x >> 5);  // 0..16383
  const int tm   = wid >> 5;                             // 0..511  (mu row tile)
  const int td   = wid & 31;                             // 0..31   (W row / out col tile)
  const int lane = threadIdx.x & 31;
  const int half = lane >> 4;
  const int l15  = lane & 15;

  const u16* arh = mu_hi + (size_t)(tm * 16 + l15) * DD;
  const u16* arl = mu_lo + (size_t)(tm * 16 + l15) * DD;
  const u16* brh = w_hi  + (size_t)(td * 16 + l15) * DD;
  const u16* brl = w_lo  + (size_t)(td * 16 + l15) * DD;

  v8f acc = {0.f, 0.f, 0.f, 0.f, 0.f, 0.f, 0.f, 0.f};
#pragma unroll
  for (int kc = 0; kc < DD / 32; ++kc) {
    const int k0 = kc * 32;
    acc = wmma3(frag_A(arh, k0, half), frag_A(arl, k0, half),
                frag_B(brh, k0, half), frag_B(brl, k0, half), acc);
  }

  const int d = td * 16 + l15;
  const float bv = bias[d];
#pragma unroll
  for (int r = 0; r < 8; ++r) {
    const int m = tm * 16 + r + 8 * half;  // C/D layout: vgpr r, lane-half -> row
    float v = tanhf(acc[r] + bv);
    u16 hb = f2bf(v);
    float hf = bf2f(hb);
    u16 lb = f2bf(v - hf);
    f_hi[(size_t)m * DD + d] = hb;
    f_lo[(size_t)m * DD + d] = lb;
  }
}

// ---- Kernel 3: row sum-of-squares from split bf16 ---------------------------
__global__ __launch_bounds__(256) void rowsq_split_kernel(
    const u16* __restrict__ hi, const u16* __restrict__ lo, float* __restrict__ sq) {
  const int row = blockIdx.x;
  const u16* ph = hi + (size_t)row * DD;
  const u16* pl = lo + (size_t)row * DD;
  float acc = 0.f;
  for (int c = threadIdx.x; c < DD; c += 256) {
    float v = bf2f(ph[c]) + bf2f(pl[c]);
    acc += v * v;
  }
  __shared__ float red[256];
  red[threadIdx.x] = acc;
  __syncthreads();
  for (int s = 128; s > 0; s >>= 1) {
    if (threadIdx.x < s) red[threadIdx.x] += red[threadIdx.x + s];
    __syncthreads();
  }
  if (threadIdx.x == 0) sq[row] = red[0];
}

// ---- Kernel 4: fused cross-GEMM + online logsumexp (flash-style) ------------
// Block = 8 waves, one 16-row x tile per block resident in LDS.
// Wave w sweeps m-tiles w, w+8, ... with split-bf16 WMMA; per 16x16 score
// tile does max-rescaled running sum-of-exp via width-16 shfl_xor reductions.
__global__ __launch_bounds__(256) void cross_lse_kernel(
    const u16* __restrict__ x_hi, const u16* __restrict__ x_lo,
    const u16* __restrict__ f_hi, const u16* __restrict__ f_lo,
    const float* __restrict__ x_sq, const float* __restrict__ f_sq,
    float* __restrict__ out) {
  constexpr int LDP = DD + 8;  // pad 16B per row to spread LDS banks
  __shared__ __align__(16) u16 xs_hi[16][LDP];
  __shared__ __align__(16) u16 xs_lo[16][LDP];
  __shared__ float wmax_s[8][16];
  __shared__ float wsum_s[8][16];

  const int nb = blockIdx.x * 16;

  // Stage the 16x512 hi/lo x tile into LDS as b128 chunks
  {
    const uint4* gh = reinterpret_cast<const uint4*>(x_hi + (size_t)nb * DD);
    const uint4* gl = reinterpret_cast<const uint4*>(x_lo + (size_t)nb * DD);
    for (int i = threadIdx.x; i < 16 * DD / 8; i += 256) {
      const int row = i >> 6;          // 512/8 = 64 chunks per row
      const int c   = (i & 63) * 8;
      *reinterpret_cast<uint4*>(&xs_hi[row][c]) = gh[i];
      *reinterpret_cast<uint4*>(&xs_lo[row][c]) = gl[i];
    }
  }
  __syncthreads();

  const int lane = threadIdx.x & 31;
  const int w    = threadIdx.x >> 5;
  const int half = lane >> 4;
  const int l15  = lane & 15;

  const u16* ah_row = &xs_hi[l15][0];
  const u16* al_row = &xs_lo[l15][0];

  float run_max[8], run_sum[8];
#pragma unroll
  for (int r = 0; r < 8; ++r) { run_max[r] = -__builtin_inff(); run_sum[r] = 0.f; }

  for (int mt = w; mt < MM / 16; mt += 8) {
    const int mb = mt * 16;
    const u16* bh_row = f_hi + (size_t)(mb + l15) * DD;
    const u16* bl_row = f_lo + (size_t)(mb + l15) * DD;

    v8f acc = {0.f, 0.f, 0.f, 0.f, 0.f, 0.f, 0.f, 0.f};
#pragma unroll
    for (int kc = 0; kc < DD / 32; ++kc) {
      const int k0 = kc * 32;
      acc = wmma3(frag_A(ah_row, k0, half), frag_A(al_row, k0, half),
                  frag_B(bh_row, k0, half), frag_B(bl_row, k0, half), acc);
    }

    const float fs = 0.5f * f_sq[mb + l15];
#pragma unroll
    for (int r = 0; r < 8; ++r) {
      const float s = acc[r] - fs;  // score; the -0.5*x_sq row constant is folded at the end
      float t = s;
      t = fmaxf(t, __shfl_xor(t, 1, 16));
      t = fmaxf(t, __shfl_xor(t, 2, 16));
      t = fmaxf(t, __shfl_xor(t, 4, 16));
      t = fmaxf(t, __shfl_xor(t, 8, 16));
      const float nm = fmaxf(run_max[r], t);
      const float scale = __expf(run_max[r] - nm);
      float p = __expf(s - nm);
      p += __shfl_xor(p, 1, 16);
      p += __shfl_xor(p, 2, 16);
      p += __shfl_xor(p, 4, 16);
      p += __shfl_xor(p, 8, 16);
      run_sum[r] = run_sum[r] * scale + p;
      run_max[r] = nm;
    }
  }

  // Cross-wave combine: lane 0 of each 16-lane group holds rows r+8*half
  if (l15 == 0) {
#pragma unroll
    for (int r = 0; r < 8; ++r) {
      wmax_s[w][r + 8 * half] = run_max[r];
      wsum_s[w][r + 8 * half] = run_sum[r];
    }
  }
  __syncthreads();

  if (threadIdx.x < 16) {
    const int n = threadIdx.x;
    float gmax = -__builtin_inff();
#pragma unroll
    for (int ww = 0; ww < 8; ++ww) gmax = fmaxf(gmax, wmax_s[ww][n]);
    float gsum = 0.f;
#pragma unroll
    for (int ww = 0; ww < 8; ++ww) gsum += wsum_s[ww][n] * __expf(wmax_s[ww][n] - gmax);
    const float lse = gmax + __logf(gsum) - 0.5f * x_sq[nb + n];
    atomicAdd(out, -lse);
  }
}

// ---- Host launch ------------------------------------------------------------
extern "C" void kernel_launch(void* const* d_in, const int* in_sizes, int n_in,
                              void* d_out, int out_size, void* d_ws, size_t ws_size,
                              hipStream_t stream) {
  const float* x  = (const float*)d_in[0];  // (N, 512)
  const float* mu = (const float*)d_in[1];  // (M, 512)
  const float* W  = (const float*)d_in[2];  // (512, 512)
  const float* b  = (const float*)d_in[3];  // (512,)
  float* out = (float*)d_out;

  char* ws = (char*)d_ws;
  size_t off = 0;
  auto carve = [&](size_t bytes) -> void* {
    void* p = ws + off;
    off += (bytes + 255) & ~(size_t)255;
    return p;
  };
  u16* x_hi  = (u16*)carve((size_t)NN * DD * 2);
  u16* x_lo  = (u16*)carve((size_t)NN * DD * 2);
  u16* mu_hi = (u16*)carve((size_t)MM * DD * 2);
  u16* mu_lo = (u16*)carve((size_t)MM * DD * 2);
  u16* w_hi  = (u16*)carve((size_t)DD * DD * 2);
  u16* w_lo  = (u16*)carve((size_t)DD * DD * 2);
  u16* f_hi  = (u16*)carve((size_t)MM * DD * 2);
  u16* f_lo  = (u16*)carve((size_t)MM * DD * 2);
  float* x_sq   = (float*)carve((size_t)NN * 4);
  float* f_sq   = (float*)carve((size_t)MM * 4);
  float* junk_sq = (float*)carve((size_t)MM * 4);

  prep_split_kernel<<<NN, 256, 0, stream>>>(x, x_hi, x_lo, x_sq);
  prep_split_kernel<<<MM, 256, 0, stream>>>(mu, mu_hi, mu_lo, junk_sq);
  prep_split_kernel<<<DD, 256, 0, stream>>>(W, w_hi, w_lo, junk_sq);

  fmu_gemm_kernel<<<(MM / 16) * (DD / 16) / 8, 256, 0, stream>>>(
      mu_hi, mu_lo, w_hi, w_lo, b, f_hi, f_lo);

  rowsq_split_kernel<<<MM, 256, 0, stream>>>(f_hi, f_lo, f_sq);

  hipMemsetAsync(d_out, 0, sizeof(float), stream);
  cross_lse_kernel<<<NN / 16, 256, 0, stream>>>(x_hi, x_lo, f_hi, f_lo, x_sq, f_sq, out);
}
